// unit_gmlp_50723563766524
// MI455X (gfx1250) — matrix operand order repaired and checked by reading, not compiled, not run on hardware
//
#include <hip/hip_runtime.h>

// ---------------------------------------------------------------------------
// Fused unit-gMLP block for MI455X (gfx1250, wave32, WMMA).
//   out = relu( conv2( spatial_mix( bn12(gelu(conv1(bn0(x)))) ) ) + b2 + x )
// One workgroup = one (batch, 4-timestep chunk). All intermediates in LDS.
// All GEMMs via v_wmma_f32_16x16x32_f16 (f16 in, f32 accumulate).
// ---------------------------------------------------------------------------

#define BB 128
#define CC 128
#define TT 64
#define VV 25
#define HH 4
#define DD 32
#define TB 4                 // timesteps per workgroup
#define PVALID (TB * VV)     // 100 valid pixels per chunk
#define PPAD 112             // padded to 7 M-tiles of 16
#define MT 7
#define EPSN 1e-5f

typedef _Float16 f16;
typedef __attribute__((ext_vector_type(4)))  _Float16 v4h;
typedef __attribute__((ext_vector_type(8)))  _Float16 v8h;
typedef __attribute__((ext_vector_type(16))) _Float16 v16h;
typedef __attribute__((ext_vector_type(4)))  float     v4f;
typedef __attribute__((ext_vector_type(8)))  float     v8f;

// ---- LDS layout (bytes). Total 242176 < 320 KB WGP budget. ----
#define OFF_W1   0            // f16[256][128]  = 65536
#define OFF_W2   65536        // f16[128][128]  = 32768
#define OFF_TOPO 98304        // f16[4][32][32] =  8192 (u,v padded 25->32)
#define OFF_SPAT 106496       // f16[4][32][32] =  8192
#define OFF_S1   114688       // float[256] bn1/bn2 scale
#define OFF_T1   115712      // float[256] bn1/bn2 bias
#define OFF_B1   116736      // float[256] conv1 bias
#define OFF_B2   117760      // float[128] conv2 bias
#define OFF_S0   118272      // float[128] bn0 scale
#define OFF_T0   118784      // float[128] bn0 bias
#define OFF_XS   119296      // f16[112][128] bn0(x), conv1 A operand = 28672
#define OFF_Z1   147968      // f16[512 rows(h,d,t)][32 v] = 32768
#define OFF_Z2   180736      // f16[512][32]               = 32768
#define OFF_YS   213504      // f16[112][128] spatial out  = 28672
#define OFF_OS   OFF_Z1      // float[112][128]=57344, aliases dead z1+z2
#define SMEM_BYTES 242176

static __device__ __forceinline__ v16h catv(v8h lo, v8h hi) {
  v16h r;
#pragma unroll
  for (int i = 0; i < 8; ++i) { r[i] = lo[i]; r[i + 8] = hi[i]; }
  return r;
}

static __device__ __forceinline__ v8f vzero8() {
  v8f z;
#pragma unroll
  for (int i = 0; i < 8; ++i) z[i] = 0.0f;
  return z;
}

// Branchless exact-GELU: 0.5*x*(1+erf(x/sqrt2)) with Abramowitz-Stegun 7.1.26
// erf approximation (|abs err| <= 1.5e-7, << f16 GEMM noise). One v_rcp_f32,
// one v_exp_f32, 5 FMAs; no divergent exec-mask path (unlike libm erff).
static __device__ __forceinline__ float gelu_fast(float x) {
  float z  = x * 0.70710678118654752f;
  float az = fabsf(z);
  float t  = __builtin_amdgcn_rcpf(1.0f + 0.3275911f * az);
  float p  = 1.061405429f;
  p = p * t - 1.453152027f;
  p = p * t + 1.421413741f;
  p = p * t - 0.284496736f;
  p = p * t + 0.254829592f;
  p = p * t;
  // exp(-az^2) = exp2(-az^2 * log2(e)); arg <= 0 so no overflow fixups needed
  float e  = __builtin_amdgcn_exp2f(-1.4426950408889634f * az * az);
  float ea = 1.0f - p * e;                 // erf(|z|)
  float er = copysignf(ea, z);             // erf(z)
  return 0.5f * x * (1.0f + er);
}

// ISA 7.12.2: 16-bit A tile (16xK32). Lane L(0-15): M=L, K{0-7,16-23};
// lane L+16: M=L, K{8-15,24-31}. Row-major [row][K] source, stride in f16.
static __device__ __forceinline__ v16h load_Atile(const f16* base, int row,
                                                  int stride, int kbase, int hi) {
  const f16* p = base + row * stride + kbase + hi * 8;
  return catv(*(const v8h*)p, *(const v8h*)(p + 16));
}

// 16-bit B tile (K32xN16). Lane n(0-15): N=n, K{0-7,16-23};
// lane n+16: N=n, K{8-15,24-31}. Row-major [n][K] source (= W[outch][inch]).
static __device__ __forceinline__ v16h load_Btile(const f16* base, int row,
                                                  int stride, int kbase, int hi) {
  const f16* p = base + row * stride + kbase + hi * 8;
  return catv(*(const v8h*)p, *(const v8h*)(p + 16));
}

__global__ __launch_bounds__(256, 1)
void unit_gmlp_fused(const float* __restrict__ x,
                     const float* __restrict__ g0, const float* __restrict__ be0,
                     const float* __restrict__ m0, const float* __restrict__ va0,
                     const float* __restrict__ w1, const float* __restrict__ cb1,
                     const float* __restrict__ g1, const float* __restrict__ be1,
                     const float* __restrict__ m1, const float* __restrict__ va1,
                     const float* __restrict__ g2, const float* __restrict__ be2,
                     const float* __restrict__ m2, const float* __restrict__ va2,
                     const float* __restrict__ topo, const float* __restrict__ spat,
                     const float* __restrict__ w2, const float* __restrict__ cb2,
                     float* __restrict__ out) {
  extern __shared__ char smem[];
  f16*   sW1 = (f16*)(smem + OFF_W1);
  f16*   sW2 = (f16*)(smem + OFF_W2);
  f16*   sTp = (f16*)(smem + OFF_TOPO);
  f16*   sSp = (f16*)(smem + OFF_SPAT);
  float* sS1 = (float*)(smem + OFF_S1);
  float* sT1 = (float*)(smem + OFF_T1);
  float* sB1 = (float*)(smem + OFF_B1);
  float* sB2 = (float*)(smem + OFF_B2);
  float* sS0 = (float*)(smem + OFF_S0);
  float* sT0 = (float*)(smem + OFF_T0);
  f16*   xs  = (f16*)(smem + OFF_XS);
  f16*   z1s = (f16*)(smem + OFF_Z1);
  f16*   z2s = (f16*)(smem + OFF_Z2);
  f16*   ys  = (f16*)(smem + OFF_YS);
  float* os  = (float*)(smem + OFF_OS);

  const int tid = threadIdx.x;
  const int w   = tid >> 5;    // wave id 0..7
  const int l   = tid & 31;    // lane id
  const int ln  = l & 15;
  const int hi  = l >> 4;
  const int b   = blockIdx.x >> 4;          // batch
  const int t0  = (blockIdx.x & 15) * TB;   // timestep base

  // ---- phase 0a: fold BN params; zero padded z buffers -------------------
  if (tid < CC) {
    float inv = g0[tid] * rsqrtf(va0[tid] + EPSN);
    sS0[tid] = inv;
    sT0[tid] = be0[tid] - m0[tid] * inv;
    sB2[tid] = cb2[tid];
  }
  {
    int o = tid;  // 0..255 = blockDim
    float gg, bb, mm, vv;
    if (o < CC) { gg = g1[o]; bb = be1[o]; mm = m1[o]; vv = va1[o]; }
    else        { int oo = o - CC; gg = g2[oo]; bb = be2[oo]; mm = m2[oo]; vv = va2[oo]; }
    float inv = gg * rsqrtf(vv + EPSN);
    sS1[o] = inv;
    sT1[o] = bb - mm * inv;
    sB1[o] = cb1[o];
  }
  {
    uint32_t* z32 = (uint32_t*)z1s;  // z1s+z2s contiguous: 65536 B
    for (int i = tid; i < 16384; i += 256) z32[i] = 0u;
  }
  __syncthreads();

  // ---- phase 0b: stage weights + bn0(x) into LDS as f16 (vectorized) -----
  {
    const v4f* w1v = (const v4f*)w1;
    v4h* d = (v4h*)sW1;
    for (int i = tid; i < (2 * CC * CC) / 4; i += 256) {
      v4f a = w1v[i];
      v4h hv;
#pragma unroll
      for (int j = 0; j < 4; ++j) hv[j] = (f16)a[j];
      d[i] = hv;
    }
  }
  {
    const v4f* w2v = (const v4f*)w2;
    v4h* d = (v4h*)sW2;
    for (int i = tid; i < (CC * CC) / 4; i += 256) {
      v4f a = w2v[i];
      v4h hv;
#pragma unroll
      for (int j = 0; j < 4; ++j) hv[j] = (f16)a[j];
      d[i] = hv;
    }
  }
  for (int i = tid; i < HH * 32 * 32; i += 256) {
    int h = i >> 10, r = (i >> 5) & 31, v = i & 31;
    f16 tv = (f16)0.0f, sv = (f16)0.0f;
    if (r < VV && v < VV) {
      int src = (h * VV + r) * VV + v;
      tv = (f16)topo[src];
      sv = (f16)spat[src];
    }
    sTp[i] = tv;
    sSp[i] = sv;
  }
  for (int i = tid; i < CC * PVALID; i += 256) {
    int c = i / PVALID, p = i - c * PVALID;
    int t = p / VV, v = p - t * VV;
    float xv = x[((b * CC + c) * TT + t0 + t) * VV + v];
    xs[p * CC + c] = (f16)(xv * sS0[c] + sT0[c]);
  }
  for (int i = tid; i < (PPAD - PVALID) * CC; i += 256)
    xs[PVALID * CC + i] = (f16)0.0f;
  // prefetch residual x lines for phase 4 (global_prefetch_b8)
  for (int i = tid * 16; i < CC * PVALID; i += 256 * 16) {
    int c = i / PVALID, p = i - c * PVALID;
    int t = p / VV, v = p - t * VV;
    __builtin_prefetch(&x[((b * CC + c) * TT + t0 + t) * VV + v], 0, 3);
  }
  __syncthreads();

  // ---- phase 1: conv1 (M=pix, N=256, K=128) + gelu + bn1/bn2 -------------
  // Wave holds B-tiles for outch groups {w, w+8} across all 4 K-steps.
  v16h B1r[2][4];
#pragma unroll
  for (int nn = 0; nn < 2; ++nn) {
    int nrow = (w + nn * 8) * 16 + ln;
#pragma unroll
    for (int ks = 0; ks < 4; ++ks)
      B1r[nn][ks] = load_Btile(sW1, nrow, CC, ks * 32, hi);
  }
#pragma unroll
  for (int m = 0; m < MT; ++m) {
    int arow = m * 16 + ln;
    v16h a[4];
#pragma unroll
    for (int ks = 0; ks < 4; ++ks)
      a[ks] = load_Atile(xs, arow, CC, ks * 32, hi);
    v8f acc0 = vzero8(), acc1 = vzero8();
#pragma unroll
    for (int ks = 0; ks < 4; ++ks) {
      acc0 = __builtin_amdgcn_wmma_f32_16x16x32_f16(false, a[ks], false, B1r[0][ks],
                                                    (short)0, acc0, false, false);
      acc1 = __builtin_amdgcn_wmma_f32_16x16x32_f16(false, a[ks], false, B1r[1][ks],
                                                    (short)0, acc1, false, false);
    }
#pragma unroll
    for (int nn = 0; nn < 2; ++nn) {
      int o = (w + nn * 8) * 16 + ln;
      float sc = sS1[o], tb = sT1[o], bi = sB1[o];
      f16* zdst = (o < CC) ? z1s : z2s;
      int  ch   = (o < CC) ? o : (o - CC);
      const v8f& acc = nn ? acc1 : acc0;
#pragma unroll
      for (int r = 0; r < 8; ++r) {
        // pix depends only on (m, r) [compile-time] and hi [0/1]: fold the
        // /25 and %25 into immediates + one select instead of div chains.
        const int pixLo = m * 16 + r;
        const int pixHi = pixLo + 8;
        if (pixLo < PVALID) {
          const int offLo = (pixLo / VV) * 32 + (pixLo % VV);
          const int offHi = (pixHi / VV) * 32 + (pixHi % VV);
          float av = acc[r] + bi;
          float zv = gelu_fast(av) * sc + tb;
          int  off   = hi ? offHi : offLo;
          bool valid = hi ? (pixHi < PVALID) : true;
          if (valid) zdst[ch * (TB * 32) + off] = (f16)zv;
        }
      }
    }
  }
  __syncthreads();

  // ---- phase 2: spatial mix, per head: (128 rows(d,t)) x (u) over K=v ----
  for (int j = w; j < 64; j += 8) {
    int h = j >> 4, rt = (j >> 1) & 7, ut = j & 1;
    int arow = h * 128 + rt * 16 + ln;
    v16h a1 = load_Atile(z1s, arow, 32, 0, hi);
    v16h a2 = load_Atile(z2s, arow, 32, 0, hi);
    int brow = h * 32 + ut * 16 + ln;
    v16h bt = load_Btile(sTp, brow, 32, 0, hi);
    v16h bs = load_Btile(sSp, brow, 32, 0, hi);
    v8f y1 = __builtin_amdgcn_wmma_f32_16x16x32_f16(false, a1, false, bt,
                                                    (short)0, vzero8(), false, false);
    v8f at = __builtin_amdgcn_wmma_f32_16x16x32_f16(false, a2, false, bs,
                                                    (short)0, vzero8(), false, false);
    int u = ut * 16 + ln;
    if (u < VV) {
#pragma unroll
      for (int r = 0; r < 8; ++r) {
        int rowm = rt * 16 + r + hi * 8;   // = d*TB + t within head
        int d = rowm >> 2, t = rowm & 3;
        float z1v = (float)z1s[((h * DD + d) * TB + t) * 32 + u];
        float yv  = y1[r] + z1v * at[r];
        int pix = t * VV + u, ch = h * DD + d;
        ys[pix * CC + ch] = (f16)yv;
      }
    }
  }
  __syncthreads();

  // ---- phase 3: conv2 (M=pix, N=128, K=128) ------------------------------
  v16h B2r[4];
  {
    int nrow = w * 16 + ln;
#pragma unroll
    for (int ks = 0; ks < 4; ++ks)
      B2r[ks] = load_Btile(sW2, nrow, CC, ks * 32, hi);
  }
#pragma unroll
  for (int m = 0; m < MT; ++m) {
    int arow = m * 16 + ln;
    v16h a[4];
#pragma unroll
    for (int ks = 0; ks < 4; ++ks)
      a[ks] = load_Atile(ys, arow, CC, ks * 32, hi);
    v8f acc = vzero8();
#pragma unroll
    for (int ks = 0; ks < 4; ++ks)
      acc = __builtin_amdgcn_wmma_f32_16x16x32_f16(false, a[ks], false, B2r[ks],
                                                   (short)0, acc, false, false);
    int o = w * 16 + ln;
#pragma unroll
    for (int r = 0; r < 8; ++r) {
      const int pixLo = m * 16 + r;
      const int pixHi = pixLo + 8;
      if (pixLo < PVALID) {
        bool valid = hi ? (pixHi < PVALID) : true;
        int  pixoff = pixLo * CC + (hi ? 8 * CC : 0);
        if (valid) os[pixoff + o] = acc[r];  // os aliases dead z1/z2
      }
    }
  }
  __syncthreads();

  // ---- phase 4: bias + residual + relu, coalesced (v-contiguous) store ---
  for (int i = tid; i < CC * PVALID; i += 256) {
    int ch = i / PVALID, p = i - ch * PVALID;
    int t = p / VV, v = p - t * VV;
    int gidx = ((b * CC + ch) * TT + t0 + t) * VV + v;
    float val = os[p * CC + ch] + sB2[ch] + x[gidx];
    out[gidx] = fmaxf(val, 0.0f);
  }
}

extern "C" void kernel_launch(void* const* d_in, const int* in_sizes, int n_in,
                              void* d_out, int out_size, void* d_ws, size_t ws_size,
                              hipStream_t stream) {
  (void)in_sizes; (void)n_in; (void)d_ws; (void)ws_size; (void)out_size;
  const float* x    = (const float*)d_in[0];
  const float* g0   = (const float*)d_in[1];
  const float* be0  = (const float*)d_in[2];
  const float* m0   = (const float*)d_in[3];
  const float* va0  = (const float*)d_in[4];
  const float* w1   = (const float*)d_in[5];
  const float* cb1  = (const float*)d_in[6];
  const float* g1   = (const float*)d_in[7];
  const float* be1  = (const float*)d_in[8];
  const float* m1   = (const float*)d_in[9];
  const float* va1  = (const float*)d_in[10];
  const float* g2   = (const float*)d_in[11];
  const float* be2  = (const float*)d_in[12];
  const float* m2   = (const float*)d_in[13];
  const float* va2  = (const float*)d_in[14];
  const float* topo = (const float*)d_in[15];
  const float* spat = (const float*)d_in[16];
  const float* w2   = (const float*)d_in[17];
  const float* cb2  = (const float*)d_in[18];
  float* out = (float*)d_out;

  dim3 grid(BB * (TT / TB));   // 128 * 16 = 2048 workgroups
  dim3 block(256);             // 8 waves (wave32)
  unit_gmlp_fused<<<grid, block, SMEM_BYTES, stream>>>(
      x, g0, be0, m0, va0, w1, cb1, g1, be1, m1, va1,
      g2, be2, m2, va2, topo, spat, w2, cb2, out);
}
